// GPool_59030030516950
// MI455X (gfx1250) — 8-bit matrix operand order repaired and checked by reading, compiled, not run on hardware
//
#include <hip/hip_runtime.h>
#include <stdint.h>

typedef int v8i __attribute__((ext_vector_type(8)));

#define N_NODES 8192
#define N_EDGES 262144
#define D_FEAT  256
#define KSEL    4096

__global__ void score_kernel(const float* __restrict__ x,
                             const float* __restrict__ w,
                             const float* __restrict__ b,
                             float* __restrict__ scores) {
    int n = blockIdx.x * blockDim.x + threadIdx.x;
    if (n >= N_NODES) return;
    const float4* xr = (const float4*)(x + (size_t)n * D_FEAT);
    const float4* wr = (const float4*)w;
    float acc = 0.f;
#pragma unroll 8
    for (int i = 0; i < D_FEAT / 4; ++i) {
        float4 a = xr[i];
        float4 c = wr[i];
        acc += a.x * c.x + a.y * c.y + a.z * c.z + a.w * c.w;
    }
    float t = acc + b[0];
    scores[n] = 1.0f / (1.0f + __expf(-t));
}

__device__ __forceinline__ unsigned int float_ord(float f) {
    unsigned int u = __float_as_uint(f);
    return (u & 0x80000000u) ? ~u : (u | 0x80000000u);
}

__global__ void build_keys_kernel(const float* __restrict__ scores,
                                  unsigned long long* __restrict__ keys) {
    int n = blockIdx.x * blockDim.x + threadIdx.x;
    if (n >= N_NODES) return;
    keys[n] = ((unsigned long long)float_ord(scores[n]) << 32) |
              (unsigned int)(~(unsigned int)n);
}

__global__ void bitonic_step_kernel(unsigned long long* __restrict__ keys,
                                    int j, int k) {
    int i = blockIdx.x * blockDim.x + threadIdx.x;
    if (i >= N_NODES) return;
    int ixj = i ^ j;
    if (ixj > i) {
        unsigned long long a = keys[i];
        unsigned long long b = keys[ixj];
        bool up = ((i & k) == 0);
        bool do_swap = up ? (a < b) : (a > b);
        if (do_swap) { keys[i] = b; keys[ixj] = a; }
    }
}

__global__ void extract_kernel(const unsigned long long* __restrict__ keys,
                               const float* __restrict__ scores,
                               int* __restrict__ idx,
                               float* __restrict__ values,
                               int* __restrict__ rank) {
    int r = blockIdx.x * blockDim.x + threadIdx.x;
    if (r >= N_NODES) return;
    unsigned long long key = keys[r];
    int n = (int)(~(unsigned int)key);
    idx[r] = n;
    values[r] = scores[n];
    rank[n] = r;
}

__global__ void gather_scale_kernel(const float* __restrict__ x,
                                    const int* __restrict__ idx,
                                    const float* __restrict__ values,
                                    float* __restrict__ out) {
    int t = blockIdx.x * blockDim.x + threadIdx.x;
    if (t >= KSEL * (D_FEAT / 4)) return;
    int r = t / (D_FEAT / 4);
    int c = t % (D_FEAT / 4);
    float v = values[r];
    float4 a = ((const float4*)(x + (size_t)idx[r] * D_FEAT))[c];
    a.x *= v; a.y *= v; a.z *= v; a.w *= v;
    ((float4*)out)[t] = a;
}

__global__ void zero16_kernel(uint4* __restrict__ p, size_t n16) {
    size_t i = (size_t)blockIdx.x * blockDim.x + threadIdx.x;
    if (i < n16) p[i] = make_uint4(0u, 0u, 0u, 0u);
}

__global__ void scatter_kernel(const int* __restrict__ ei,
                               const int* __restrict__ rank,
                               unsigned char* __restrict__ A,
                               unsigned char* __restrict__ BT) {
    int e = blockIdx.x * blockDim.x + threadIdx.x;
    if (e >= N_EDGES) return;
    int s = ei[e];
    int d = ei[N_EDGES + e];
    int rs = rank[s];
    int rd = rank[d];
    if (rs < KSEL) A [(size_t)rs * N_NODES + d] = 1;
    if (rd < KSEL) BT[(size_t)rd * N_NODES + s] = 1;
}

__global__ __launch_bounds__(256) void bmm_mask_kernel(
        const unsigned char* __restrict__ A,
        const unsigned char* __restrict__ BT,
        float* __restrict__ mask_out) {
    const int lane = threadIdx.x & 31;
    const int wave = threadIdx.x >> 5;

    const int wi = blockIdx.x >> 4;
    const int wj = ((blockIdx.x & 15) << 3) | wave;
    const int i0 = wi * 64;
    const int j0 = wj * 32;

    const int h  = lane >> 4;
    const int mr = lane & 15;

    const unsigned char* arow0 = A  + (size_t)(i0 + mr) * N_NODES;
    const unsigned char* arow1 = arow0 + (size_t)16 * N_NODES;
    const unsigned char* arow2 = arow0 + (size_t)32 * N_NODES;
    const unsigned char* arow3 = arow0 + (size_t)48 * N_NODES;
    const unsigned char* brow0 = BT + (size_t)(j0 + mr) * N_NODES;
    const unsigned char* brow1 = brow0 + (size_t)16 * N_NODES;

    v8i acc00 = {}, acc01 = {};
    v8i acc10 = {}, acc11 = {};
    v8i acc20 = {}, acc21 = {};
    v8i acc30 = {}, acc31 = {};

    for (int k0 = 0; k0 < N_NODES; k0 += 64) {
        v8i a0, a1, a2, a3, b0, b1;
#pragma unroll
        for (int v = 0; v < 8; ++v) {
            int ka = k0 + ((v >> 1) << 4) + ((v & 1) << 2) + (h << 3);
            int kb = k0 + ((v >> 1) << 4) + ((v & 1) << 2) + (h << 3);
            a0[v] = *(const int*)(arow0 + ka);
            a1[v] = *(const int*)(arow1 + ka);
            a2[v] = *(const int*)(arow2 + ka);
            a3[v] = *(const int*)(arow3 + ka);
            b0[v] = *(const int*)(brow0 + kb);
            b1[v] = *(const int*)(brow1 + kb);
        }
        acc00 = __builtin_amdgcn_wmma_i32_16x16x64_iu8(false, a0, false, b0,
                                                       acc00, false, false);
        acc10 = __builtin_amdgcn_wmma_i32_16x16x64_iu8(false, a1, false, b0,
                                                       acc10, false, false);
        acc20 = __builtin_amdgcn_wmma_i32_16x16x64_iu8(false, a2, false, b0,
                                                       acc20, false, false);
        acc30 = __builtin_amdgcn_wmma_i32_16x16x64_iu8(false, a3, false, b0,
                                                       acc30, false, false);
        acc01 = __builtin_amdgcn_wmma_i32_16x16x64_iu8(false, a0, false, b1,
                                                       acc01, false, false);
        acc11 = __builtin_amdgcn_wmma_i32_16x16x64_iu8(false, a1, false, b1,
                                                       acc11, false, false);
        acc21 = __builtin_amdgcn_wmma_i32_16x16x64_iu8(false, a2, false, b1,
                                                       acc21, false, false);
        acc31 = __builtin_amdgcn_wmma_i32_16x16x64_iu8(false, a3, false, b1,
                                                       acc31, false, false);
    }

#pragma unroll
    for (int t = 0; t < 8; ++t) {
        v8i acc = (t == 0) ? acc00 : (t == 1) ? acc01 :
                  (t == 2) ? acc10 : (t == 3) ? acc11 :
                  (t == 4) ? acc20 : (t == 5) ? acc21 :
                  (t == 6) ? acc30 : acc31;
        int ib = i0 + ((t >> 1) << 4);
        int jb = j0 + ((t & 1) << 4);
#pragma unroll
        for (int r = 0; r < 8; ++r) {
            int gi = ib + r + 8 * h;
            int gj = jb + mr;
            float val = ((gi > gj) && (acc[r] > 0)) ? 1.0f : 0.0f;
            mask_out[(size_t)gi * KSEL + gj] = val;
        }
    }
}

extern "C" void kernel_launch(void* const* d_in, const int* in_sizes, int n_in,
                              void* d_out, int out_size, void* d_ws, size_t ws_size,
                              hipStream_t stream) {
    const float* x  = (const float*)d_in[0];
    const int*   ei = (const int*)d_in[1];
    const float* pw = (const float*)d_in[2];
    const float* pb = (const float*)d_in[3];

    float* out   = (float*)d_out;
    float* new_h = out;
    float* mask  = out + (size_t)KSEL * D_FEAT;

    char* ws = (char*)d_ws;
    float* scores = (float*)ws;                       ws += (size_t)N_NODES * 4;
    unsigned long long* keys = (unsigned long long*)ws; ws += (size_t)N_NODES * 8;
    int*   idx    = (int*)ws;                         ws += (size_t)N_NODES * 4;
    float* values = (float*)ws;                       ws += (size_t)N_NODES * 4;
    int*   rank   = (int*)ws;                         ws += (size_t)N_NODES * 4;
    uintptr_t up = ((uintptr_t)ws + 255) & ~(uintptr_t)255;
    unsigned char* A  = (unsigned char*)up;
    unsigned char* BT = A + (size_t)KSEL * N_NODES;

    score_kernel<<<N_NODES / 256, 256, 0, stream>>>(x, pw, pb, scores);

    build_keys_kernel<<<N_NODES / 256, 256, 0, stream>>>(scores, keys);
    for (int k = 2; k <= N_NODES; k <<= 1)
        for (int j = k >> 1; j > 0; j >>= 1)
            bitonic_step_kernel<<<N_NODES / 256, 256, 0, stream>>>(keys, j, k);
    extract_kernel<<<N_NODES / 256, 256, 0, stream>>>(keys, scores, idx, values, rank);

    gather_scale_kernel<<<(KSEL * (D_FEAT / 4)) / 256, 256, 0, stream>>>(
        x, idx, values, new_h);

    size_t n16 = (size_t)2 * KSEL * N_NODES / 16;
    zero16_kernel<<<(unsigned)((n16 + 255) / 256), 256, 0, stream>>>((uint4*)A, n16);

    scatter_kernel<<<N_EDGES / 256, 256, 0, stream>>>(ei, rank, A, BT);

    bmm_mask_kernel<<<(64 * 128) / 8, 256, 0, stream>>>(A, BT, mask);
}
